// ActionConditionedTransition_56753697849969
// MI455X (gfx1250) — compile-verified, matrix operand order checked by reading, not run on hardware
//
#include <hip/hip_runtime.h>

typedef __attribute__((ext_vector_type(16))) _Float16 v16h;
typedef __attribute__((ext_vector_type(8)))  _Float16 v8h;
typedef __attribute__((ext_vector_type(4)))  _Float16 v4h;
typedef __attribute__((ext_vector_type(8)))  float    v8f;

#define BATCH 1024
#define DDIM  512
#define NACT  18

// ---------------- kernel 1: zero the per-action counters ----------------
__global__ void act_zero_counts(int* __restrict__ counts) {
    if (threadIdx.x < NACT) counts[threadIdx.x] = 0;
}

// ---------------- kernel 2: compact sample indices by action ----------------
__global__ void act_compact(const int* __restrict__ action,
                            int* __restrict__ counts,
                            int* __restrict__ idxbuf) {
    int b = blockIdx.x * blockDim.x + threadIdx.x;
    if (b < BATCH) {
        int a = action[b];
        int slot = atomicAdd(&counts[a], 1);
        idxbuf[a * BATCH + slot] = b;
    }
}

// ---------------- kernel 3: gather + f32->f16 convert state rows, grouped by action,
//                  zero-padded to a multiple of 16 rows per action ----------------
// grid = (BATCH slots, NACT); block = 128 (4 elements / thread)
__global__ __launch_bounds__(128) void act_gather(const float* __restrict__ state,
                                                  const int* __restrict__ counts,
                                                  const int* __restrict__ idxbuf,
                                                  _Float16* __restrict__ Sg) {
    const int slot = blockIdx.x;
    const int a    = blockIdx.y;
    const int cnt  = counts[a];
    const int pad  = (cnt + 15) & ~15;
    if (slot >= pad) return;

    v4h* dst = (v4h*)(Sg + ((size_t)a * BATCH + slot) * DDIM);
    const int t = threadIdx.x;
    if (slot < cnt) {
        const float* src = state + (size_t)idxbuf[a * BATCH + slot] * DDIM;
        float4 v = ((const float4*)src)[t];
        v4h h;
        h[0] = (_Float16)v.x; h[1] = (_Float16)v.y;
        h[2] = (_Float16)v.z; h[3] = (_Float16)v.w;
        dst[t] = h;
    } else {
        dst[t] = (v4h){};                           // zero padding rows
    }
}

// ---------------- kernel 4: row softmax of all 18 matrices, transposed f16 out ----
// grid = (512 rows, 18 actions), block = 256. PT[a][j*512 + i] = softmax(mat[a][i])[j]
__global__ __launch_bounds__(256) void act_softmax(const float* __restrict__ mats,
                                                   _Float16* __restrict__ PT) {
    const int row = blockIdx.x;
    const int a   = blockIdx.y;
    const int tid = threadIdx.x;
    const float* src = mats + ((size_t)a * DDIM + row) * DDIM;

    float2 v = ((const float2*)src)[tid];          // 2 elements per thread

    __shared__ float red[256];
    red[tid] = fmaxf(v.x, v.y);
    __syncthreads();
    for (int s = 128; s > 0; s >>= 1) {
        if (tid < s) red[tid] = fmaxf(red[tid], red[tid + s]);
        __syncthreads();
    }
    float mx = red[0];
    __syncthreads();

    float e0 = __expf(v.x - mx);
    float e1 = __expf(v.y - mx);
    red[tid] = e0 + e1;
    __syncthreads();
    for (int s = 128; s > 0; s >>= 1) {
        if (tid < s) red[tid] += red[tid + s];
        __syncthreads();
    }
    float inv = 1.0f / red[0];

    // transposed store: column index j = 2*tid, 2*tid+1 ; i = row
    _Float16* dst = PT + (size_t)a * DDIM * DDIM + row;
    int j = tid * 2;
    dst[(size_t)j * DDIM]       = (_Float16)(e0 * inv);
    dst[(size_t)(j + 1) * DDIM] = (_Float16)(e1 * inv);
}

// ---------------- kernel 5: grouped GEMM via WMMA f16 -> f32 ----------------
// grid = (DDIM/128, BATCH/16, NACT); block = 256 (8 waves, one 16x16 C-tile each)
__global__ __launch_bounds__(256) void act_gemm(const _Float16* __restrict__ Sg,
                                                const _Float16* __restrict__ PT,
                                                const int* __restrict__ counts,
                                                const int* __restrict__ idxbuf,
                                                float* __restrict__ out) {
    const int a   = blockIdx.z;
    const int cnt = counts[a];
    const int m0  = blockIdx.y * 16;
    if (m0 >= cnt) return;                         // uniform per block

    const int lane = threadIdx.x & 31;
    const int wave = threadIdx.x >> 5;
    const int n0   = (blockIdx.x * 8 + wave) * 16; // 16 output columns per wave
    const int col  = lane & 15;                    // N for B/C, M-row for A
    const int hi   = lane >> 4;                    // lane half
    const int c    = hi * 8;                       // K sub-offset per CDNA5 layout

    // lane's A row (zero-padded in Sg, so no masking needed in the k-loop)
    const _Float16* arow = Sg + ((size_t)a * BATCH + (m0 + col)) * DDIM;
    const _Float16* bcol = PT + ((size_t)a * DDIM + (n0 + col)) * DDIM;

    // row index for scatter (valid only when slot < cnt; shuffled in epilogue)
    const int slotA = m0 + col;
    const int ridA  = (slotA < cnt) ? idxbuf[a * BATCH + slotA] : 0;

    v8f acc = {};
    #pragma unroll 4
    for (int k = 0; k < DDIM; k += 32) {
        v8h a0 = *(const v8h*)(arow + k + c);
        v8h a1 = *(const v8h*)(arow + k + c + 16);
        v8h b0 = *(const v8h*)(bcol + k + c);
        v8h b1 = *(const v8h*)(bcol + k + c + 16);
        v16h af = __builtin_shufflevector(a0, a1,
                    0,1,2,3,4,5,6,7,8,9,10,11,12,13,14,15);
        v16h bf = __builtin_shufflevector(b0, b1,
                    0,1,2,3,4,5,6,7,8,9,10,11,12,13,14,15);
        acc = __builtin_amdgcn_wmma_f32_16x16x32_f16(
                  false, af, false, bf, (short)0, acc, false, false);
    }

    // ---- scatter C tile back to original sample rows (rids via lane shuffle) ----
    #pragma unroll
    for (int r = 0; r < 8; ++r) {
        const int M    = r + hi * 8;               // C layout: VGPR r, lane half
        const int slot = m0 + M;
        const int rid  = __shfl(ridA, M);          // lane M holds rid for slot m0+M
        if (slot < cnt) {
            out[(size_t)rid * DDIM + n0 + col] = acc[r];
        }
    }
}

// ---------------- host-side launch ----------------
extern "C" void kernel_launch(void* const* d_in, const int* in_sizes, int n_in,
                              void* d_out, int out_size, void* d_ws, size_t ws_size,
                              hipStream_t stream) {
    const float* state  = (const float*)d_in[0];
    const int*   action = (const int*)d_in[1];
    const float* mats   = (const float*)d_in[2];
    float*       out    = (float*)d_out;

    char* ws = (char*)d_ws;
    const size_t PT_BYTES = (size_t)NACT * DDIM * DDIM * sizeof(_Float16);  // 9.44 MB
    const size_t SG_BYTES = (size_t)NACT * BATCH * DDIM * sizeof(_Float16); // 18.9 MB
    _Float16* PT     = (_Float16*)ws;
    _Float16* Sg     = (_Float16*)(ws + PT_BYTES);
    int*      counts = (int*)(ws + PT_BYTES + SG_BYTES);
    int*      idxbuf = (int*)(ws + PT_BYTES + SG_BYTES + 256);

    act_zero_counts<<<1, 32, 0, stream>>>(counts);
    act_compact<<<(BATCH + 255) / 256, 256, 0, stream>>>(action, counts, idxbuf);
    act_gather<<<dim3(BATCH, NACT), 128, 0, stream>>>(state, counts, idxbuf, Sg);
    act_softmax<<<dim3(DDIM, NACT), 256, 0, stream>>>(mats, PT);
    act_gemm<<<dim3(DDIM / 128, BATCH / 16, NACT), 256, 0, stream>>>(
        Sg, PT, counts, idxbuf, out);
}
